// TernaryLinear_86380382257477
// MI455X (gfx1250) — 8-bit matrix operand order repaired and checked by reading, compiled, not run on hardware
//
#include <hip/hip_runtime.h>
#include <hip/hip_bf16.h>


typedef __attribute__((ext_vector_type(8))) int v8i;

#define TL_K 4096
#define TL_N 4096

#define WS_SUMS_OFF   0
#define WS_A_OFF      256

#define TL_LDS_STRIDE 80

__global__ void tl_init_kernel(float* sums) {
    if (threadIdx.x == 0) { sums[0] = 0.0f; sums[1] = 0.0f; }
}

__global__ __launch_bounds__(256)
void tl_reduce_abs_kernel(const float* __restrict__ p, long long n4, float* out) {
    long long i      = (long long)blockIdx.x * blockDim.x + threadIdx.x;
    long long stride = (long long)gridDim.x * blockDim.x;
    float s = 0.0f;
    const float4* p4 = (const float4*)p;
    for (; i < n4; i += stride) {
        float4 v = p4[i];
        s += fabsf(v.x) + fabsf(v.y) + fabsf(v.z) + fabsf(v.w);
    }
    #pragma unroll
    for (int off = 16; off > 0; off >>= 1) s += __shfl_down(s, off, 32);
    __shared__ float ls[8];
    int lane = threadIdx.x & 31, wave = threadIdx.x >> 5;
    if (lane == 0) ls[wave] = s;
    __syncthreads();
    if (wave == 0) {
        s = (lane < 8) ? ls[lane] : 0.0f;
        #pragma unroll
        for (int off = 4; off > 0; off >>= 1) s += __shfl_down(s, off, 32);
        if (lane == 0) atomicAdd(out, s);
    }
}

__device__ __forceinline__ int tl_tern_byte(float v, float delta) {
    signed char q = (v > delta) ? 1 : ((v < -delta) ? -1 : 0);
    return (int)(unsigned char)q;
}

__global__ __launch_bounds__(256)
void tl_quant_kernel(const float* __restrict__ src, int* __restrict__ dst8,
                     long long n4, const float* __restrict__ sums,
                     int sum_idx, float thresh_scale, float inv_count) {
    float delta = thresh_scale * (sums[sum_idx] * inv_count);
    long long i = (long long)blockIdx.x * blockDim.x + threadIdx.x;
    if (i >= n4) return;
    float4 v = ((const float4*)src)[i];
    int w = tl_tern_byte(v.x, delta)
          | (tl_tern_byte(v.y, delta) << 8)
          | (tl_tern_byte(v.z, delta) << 16)
          | (tl_tern_byte(v.w, delta) << 24);
    dst8[i] = w;
}

__device__ __forceinline__ void tl_wait_async0() {
#if __has_builtin(__builtin_amdgcn_s_wait_asynccnt)
    __builtin_amdgcn_s_wait_asynccnt(0);
#else
    asm volatile("s_wait_asynccnt 0" ::: "memory");
#endif
}

__device__ __forceinline__ void tl_async_b128(unsigned lds_off, unsigned voff,
                                              const signed char* base) {
    asm volatile("global_load_async_to_lds_b128 %0, %1, %2"
                 :: "v"(lds_off), "v"(voff), "s"(base)
                 : "memory");
}

__global__ __launch_bounds__(256)
void tl_gemm_iu8_kernel(const signed char* __restrict__ A,
                        const signed char* __restrict__ B,
                        float* __restrict__ C,
                        const float* __restrict__ sums) {
    __shared__ __align__(16) char sA[2][64 * TL_LDS_STRIDE];
    __shared__ __align__(16) char sB[2][128 * TL_LDS_STRIDE];

    const float alpha = sums[1] * (1.0f / ((float)TL_N * (float)TL_K));

    const int tid  = threadIdx.x;
    const int lane = tid & 31;
    const int wave = tid >> 5;
    const int lm = lane & 15;
    const int lh = lane >> 4;

    const int m_blk = (int)blockIdx.y * 64;
    const int n_blk = (int)blockIdx.x * 128;

    const int crow = tid >> 2;
    const int ccol = (tid & 3) * 16;
    const unsigned aVoff  = (unsigned)((m_blk + crow) * TL_K + ccol);
    const unsigned bVoff0 = (unsigned)((n_blk + crow) * TL_K + ccol);
    const unsigned bVoff1 = (unsigned)((n_blk + crow + 64) * TL_K + ccol);
    const unsigned aLoff  = (unsigned)(crow * TL_LDS_STRIDE + ccol);
    const unsigned bLoff0 = aLoff;
    const unsigned bLoff1 = (unsigned)((crow + 64) * TL_LDS_STRIDE + ccol);

    const int rA0 = ((wave >> 2) * 32 + lm) * TL_LDS_STRIDE;
    const int rA1 = rA0 + 16 * TL_LDS_STRIDE;
    const int rB0 = ((wave & 3) * 32 + lm) * TL_LDS_STRIDE;
    const int rB1 = rB0 + 16 * TL_LDS_STRIDE;

    v8i acc00 = {}, acc01 = {}, acc10 = {}, acc11 = {};

    auto prefetch = [&](int buf, int k0) {
        unsigned la = (unsigned)(unsigned long long)(&sA[buf][0]);
        unsigned lb = (unsigned)(unsigned long long)(&sB[buf][0]);
        tl_async_b128(la + aLoff,  aVoff  + (unsigned)k0, A);
        tl_async_b128(lb + bLoff0, bVoff0 + (unsigned)k0, B);
        tl_async_b128(lb + bLoff1, bVoff1 + (unsigned)k0, B);
    };

    prefetch(0, 0);

    const int KSTEPS = TL_K / 64;
    for (int i = 0; i < KSTEPS; ++i) {
        tl_wait_async0();
        __syncthreads();

        if (i + 1 < KSTEPS) prefetch((i + 1) & 1, (i + 1) * 64);

        const char* pa = &sA[i & 1][0];
        const char* pb = &sB[i & 1][0];

        v8i a0, a1, b0, b1;
        #pragma unroll
        for (int p = 0; p < 4; ++p) {
            int2 d0 = *(const int2*)(pa + rA0 + (p >> 1) * 32 + lh * 16 + (p & 1) * 8);
            int2 d1 = *(const int2*)(pa + rA1 + (p >> 1) * 32 + lh * 16 + (p & 1) * 8);
            a0[2 * p] = d0.x; a0[2 * p + 1] = d0.y;
            a1[2 * p] = d1.x; a1[2 * p + 1] = d1.y;
        }
        #pragma unroll
        for (int q = 0; q < 2; ++q) {
            int4 e0 = *(const int4*)(pb + rB0 + q * 32 + lh * 16);
            int4 e1 = *(const int4*)(pb + rB1 + q * 32 + lh * 16);
            b0[4 * q] = e0.x; b0[4 * q + 1] = e0.y; b0[4 * q + 2] = e0.z; b0[4 * q + 3] = e0.w;
            b1[4 * q] = e1.x; b1[4 * q + 1] = e1.y; b1[4 * q + 2] = e1.z; b1[4 * q + 3] = e1.w;
        }

        acc00 = __builtin_amdgcn_wmma_i32_16x16x64_iu8(true, a0, true, b0, acc00, false, false);
        acc01 = __builtin_amdgcn_wmma_i32_16x16x64_iu8(true, a0, true, b1, acc01, false, false);
        acc10 = __builtin_amdgcn_wmma_i32_16x16x64_iu8(true, a1, true, b0, acc10, false, false);
        acc11 = __builtin_amdgcn_wmma_i32_16x16x64_iu8(true, a1, true, b1, acc11, false, false);
    }

    const int m_w  = m_blk + (wave >> 2) * 32;
    const int n_w  = n_blk + (wave & 3) * 32;
    const int row0 = m_w + lh * 8;
    const int col0 = n_w + lm;
    #pragma unroll
    for (int e = 0; e < 8; ++e) {
        float* c0 = C + (long long)(row0 + e) * TL_N;
        float* c1 = C + (long long)(row0 + 16 + e) * TL_N;
        c0[col0]      = alpha * (float)acc00[e];
        c0[col0 + 16] = alpha * (float)acc01[e];
        c1[col0]      = alpha * (float)acc10[e];
        c1[col0 + 16] = alpha * (float)acc11[e];
    }
}

extern "C" void kernel_launch(void* const* d_in, const int* in_sizes, int n_in,
                              void* d_out, int out_size, void* d_ws, size_t ws_size,
                              hipStream_t stream) {
    const float* x = (const float*)d_in[0];
    const float* w = (const float*)d_in[1];
    float* out = (float*)d_out;

    const long long xN = in_sizes[0];
    const long long wN = in_sizes[1];
    const int M = (int)(xN / TL_K);
    const int N = (int)(wN / TL_K);

    float*       sums = (float*)((char*)d_ws + WS_SUMS_OFF);
    signed char* A8   = (signed char*)((char*)d_ws + WS_A_OFF);
    signed char* B8   = A8 + xN;

    tl_init_kernel<<<1, 32, 0, stream>>>(sums);

    const long long x4 = xN / 4, w4 = wN / 4;
    tl_reduce_abs_kernel<<<2048, 256, 0, stream>>>(x, x4, &sums[0]);
    tl_reduce_abs_kernel<<<2048, 256, 0, stream>>>(w, w4, &sums[1]);

    tl_quant_kernel<<<(unsigned)((x4 + 255) / 256), 256, 0, stream>>>(
        x, (int*)A8, x4, sums, 0, 0.1f, 1.0f / (float)xN);
    tl_quant_kernel<<<(unsigned)((w4 + 255) / 256), 256, 0, stream>>>(
        w, (int*)B8, w4, sums, 1, 0.05f, 1.0f / (float)wN);

    dim3 grid((unsigned)(N / 128), (unsigned)(M / 64));
    tl_gemm_iu8_kernel<<<grid, 256, 0, stream>>>(A8, B8, out, sums);
}
